// QNetworkWithValuebuffer_3358664426135
// MI455X (gfx1250) — matrix operand order repaired and checked by reading, not compiled, not run on hardware
//
#include <hip/hip_runtime.h>

#define B_     512
#define CAP_   50000
#define CAPP_  51200   // padded to 25 chunks * 2048 columns
#define D_     256
#define A_     18
#define K_     5

typedef __attribute__((ext_vector_type(8)))  __bf16 v8bf;
typedef __attribute__((ext_vector_type(16))) __bf16 v16bf;
typedef __attribute__((ext_vector_type(8)))  float  v8f;

#define BIGF 1e30f

// ---------------------------------------------------------------------------
// 1) buf_emb f32 -> bf16 + row squared norms
// ---------------------------------------------------------------------------
__global__ __launch_bounds__(256) void emb_convert_kernel(
    const float* __restrict__ emb, __bf16* __restrict__ ebf,
    float* __restrict__ bnorm)
{
    __shared__ float red[256];
    const int r = blockIdx.x;
    const int t = threadIdx.x;
    const float v = emb[(long)r * D_ + t];
    ebf[(long)r * D_ + t] = (__bf16)v;
    red[t] = v * v;
    __syncthreads();
    for (int s = 128; s > 0; s >>= 1) {
        if (t < s) red[t] += red[t + s];
        __syncthreads();
    }
    if (t == 0) bnorm[r] = red[0];
}

// ---------------------------------------------------------------------------
// 2) conv1: [512,4,84,84] -> [512,32,20,20], 8x8 stride 4, relu
// ---------------------------------------------------------------------------
__global__ __launch_bounds__(256) void conv1_kernel(
    const float* __restrict__ x, const float* __restrict__ w,
    const float* __restrict__ bias, float* __restrict__ out)
{
    const int idx = blockIdx.x * 256 + threadIdx.x;
    if (idx >= 512 * 32 * 20 * 20) return;
    const int xo = idx % 20;
    int tmp = idx / 20;
    const int yo = tmp % 20; tmp /= 20;
    const int o  = tmp % 32;
    const int b  = tmp / 32;
    float sum = bias[o];
    for (int c = 0; c < 4; ++c) {
        const float* xp = x + (((long)(b * 4 + c) * 84 + yo * 4) * 84 + xo * 4);
        const float* wp = w + ((o * 4 + c) * 64);
        #pragma unroll
        for (int ky = 0; ky < 8; ++ky)
            #pragma unroll
            for (int kx = 0; kx < 8; ++kx)
                sum = fmaf(xp[ky * 84 + kx], wp[ky * 8 + kx], sum);
    }
    out[idx] = fmaxf(sum, 0.0f);
}

// ---------------------------------------------------------------------------
// 3) conv2: [512,32,20,20] -> [512,64,9,9], 4x4 stride 2, relu
// ---------------------------------------------------------------------------
__global__ __launch_bounds__(256) void conv2_kernel(
    const float* __restrict__ x, const float* __restrict__ w,
    const float* __restrict__ bias, float* __restrict__ out)
{
    const int idx = blockIdx.x * 256 + threadIdx.x;
    if (idx >= 512 * 64 * 9 * 9) return;
    const int xo = idx % 9;
    int tmp = idx / 9;
    const int yo = tmp % 9; tmp /= 9;
    const int o  = tmp % 64;
    const int b  = tmp / 64;
    float sum = bias[o];
    for (int c = 0; c < 32; ++c) {
        const float* xp = x + (((long)(b * 32 + c) * 20 + yo * 2) * 20 + xo * 2);
        const float* wp = w + ((o * 32 + c) * 16);
        #pragma unroll
        for (int ky = 0; ky < 4; ++ky)
            #pragma unroll
            for (int kx = 0; kx < 4; ++kx)
                sum = fmaf(xp[ky * 20 + kx], wp[ky * 4 + kx], sum);
    }
    out[idx] = fmaxf(sum, 0.0f);
}

// ---------------------------------------------------------------------------
// 4) conv3: [512,64,9,9] -> [512,64,7,7], 3x3 stride 1, relu
// ---------------------------------------------------------------------------
__global__ __launch_bounds__(256) void conv3_kernel(
    const float* __restrict__ x, const float* __restrict__ w,
    const float* __restrict__ bias, float* __restrict__ out)
{
    const int idx = blockIdx.x * 256 + threadIdx.x;
    if (idx >= 512 * 64 * 7 * 7) return;
    const int xo = idx % 7;
    int tmp = idx / 7;
    const int yo = tmp % 7; tmp /= 7;
    const int o  = tmp % 64;
    const int b  = tmp / 64;
    float sum = bias[o];
    for (int c = 0; c < 64; ++c) {
        const float* xp = x + (((long)(b * 64 + c) * 9 + yo) * 9 + xo);
        const float* wp = w + ((o * 64 + c) * 9);
        #pragma unroll
        for (int ky = 0; ky < 3; ++ky)
            #pragma unroll
            for (int kx = 0; kx < 3; ++kx)
                sum = fmaf(xp[ky * 9 + kx], wp[ky * 3 + kx], sum);
    }
    out[idx] = fmaxf(sum, 0.0f);
}

// ---------------------------------------------------------------------------
// 5) fc: [512,3136] @ [3136,256] + relu -> h (f32 ws + bf16 ws + d_out copy)
// ---------------------------------------------------------------------------
__global__ __launch_bounds__(256) void fc_kernel(
    const float* __restrict__ h3, const float* __restrict__ fw,
    const float* __restrict__ fb, float* __restrict__ hf,
    __bf16* __restrict__ hb, float* __restrict__ h_out)
{
    const int idx = blockIdx.x * 256 + threadIdx.x;  // exactly 512*256 threads
    const int d = idx & 255;
    const int b = idx >> 8;
    float sum = fb[d];
    const float* xr = h3 + (long)b * 3136;
    for (int i = 0; i < 3136; ++i)
        sum = fmaf(xr[i], fw[(long)i * D_ + d], sum);
    const float v = fmaxf(sum, 0.0f);
    hf[idx] = v;
    hb[idx] = (__bf16)v;
    h_out[idx] = v;
}

// ---------------------------------------------------------------------------
// 6) parametric q: [512,256] @ [256,18] + bias
// ---------------------------------------------------------------------------
__global__ __launch_bounds__(256) void qpar_kernel(
    const float* __restrict__ h, const float* __restrict__ ow,
    const float* __restrict__ ob, float* __restrict__ q)
{
    const int idx = blockIdx.x * 256 + threadIdx.x;  // exactly 512*18 threads
    if (idx >= B_ * A_) return;
    const int a = idx % A_;
    const int b = idx / A_;
    float sum = ob[a];
    const float* hr = h + (long)b * D_;
    for (int d = 0; d < D_; ++d)
        sum = fmaf(hr[d], ow[d * A_ + a], sum);
    q[idx] = sum;
}

// ---------------------------------------------------------------------------
// 7) WMMA distance GEMM + fused per-chunk top-5
//    grid = (32 M-tiles, 25 CAP-chunks), block = 128 (4 wave32s)
//    wave w handles columns [chunk*2048 + w*512, +512)
//    ranking key: ||e||^2 - 2 h.e  (||h||^2 constant per row -> dropped)
//    emb buffer padded to CAPP_ rows: no clamping, pure pointer increments;
//    garbage columns (n >= CAP_) only poison their own C column, never inserted.
// ---------------------------------------------------------------------------
__global__ __launch_bounds__(128) void dist_topk_kernel(
    const __bf16* __restrict__ hbf, const __bf16* __restrict__ ebf,
    const float* __restrict__ bnorm,
    float* __restrict__ pd, int* __restrict__ pi)
{
    __shared__ __bf16 hlds[16 * D_];            // 8 KB   : A tile (16 x 256)
    __shared__ float  md[4 * 32 * 8 * K_];      // 20 KB  : per-lane candidates
    __shared__ int    mi[4 * 32 * 8 * K_];      // 20 KB

    const int mtile = blockIdx.x;
    const int chunk = blockIdx.y;
    const int tid   = threadIdx.x;
    const int wave  = tid >> 5;
    const int lane  = tid & 31;
    const int laneHi = lane >> 4;               // 0 / 1 half-wave
    const int m16    = lane & 15;

    // stage h tile into LDS (16 rows x 256 bf16 = 512 x float4)
    {
        const float4* src = (const float4*)(hbf + (long)mtile * 16 * D_);
        float4* dst = (float4*)hlds;
        for (int i = tid; i < 512; i += 128) dst[i] = src[i];
    }
    __syncthreads();

    // Preload all 8 A fragments (K=256 in 8 slices of 32) into registers.
    // A layout (16x32 bf16): lanes0-15 K=0..15, lanes16-31 K=16..31 (contiguous, matches B).
    v16bf afrag[8];
    #pragma unroll
    for (int ks = 0; ks < 8; ++ks) {
        const int k0 = ks * 32 + laneHi * 16;
        const v8bf alo = *(const v8bf*)(hlds + m16 * D_ + k0);
        const v8bf ahi = *(const v8bf*)(hlds + m16 * D_ + k0 + 8);
        #pragma unroll
        for (int i = 0; i < 8; ++i) { afrag[ks][i] = alo[i]; afrag[ks][i + 8] = ahi[i]; }
    }

    // per-lane sorted top-5 for each of this lane's 8 accumulator rows
    float bd[8][K_];
    int   bi[8][K_];
    #pragma unroll
    for (int r = 0; r < 8; ++r)
        #pragma unroll
        for (int j = 0; j < K_; ++j) { bd[r][j] = BIGF; bi[r][j] = 0; }

    const int colBase = chunk * 2048 + wave * 512;
    const int n0 = colBase + m16;               // this lane's first C/D column
    const __bf16* brow = ebf + (long)n0 * D_ + laneHi * 16;

    for (int t = 0; t < 32; ++t) {
        const int n = n0 + t * 16;
        const float bn = bnorm[n];              // padded buffer: always in range

        // prefetch next tile's B row (stays inside the padded ws region)
        __builtin_prefetch(brow + 16 * D_, 0, 0);

        // Two independent accumulator chains (even/odd K slices) so
        // consecutive WMMAs have no register dependency -> no hazard NOPs.
        v8f acc0 = {}, acc1 = {};
        #pragma unroll
        for (int ks = 0; ks < 4; ++ks) {
            // B fragment (32x16 bf16): lanes0-15 K=0..15, lanes16-31 K=16..31,
            // contiguous 32B per lane -> single vector load into the WMMA quad
            const v16bf bf0 = *(const v16bf*)(brow + (2 * ks)     * 32);
            const v16bf bf1 = *(const v16bf*)(brow + (2 * ks + 1) * 32);
            acc0 = __builtin_amdgcn_wmma_f32_16x16x32_bf16(
                false, afrag[2 * ks],     false, bf0, (short)0, acc0, false, false);
            acc1 = __builtin_amdgcn_wmma_f32_16x16x32_bf16(
                false, afrag[2 * ks + 1], false, bf1, (short)0, acc1, false, false);
        }

        if (n < CAP_) {
            #pragma unroll
            for (int r = 0; r < 8; ++r) {
                float dv = bn - 2.0f * (acc0[r] + acc1[r]);
                if (dv < bd[r][K_ - 1]) {        // cheap reject vs current 5th-best
                    int di = n;
                    #pragma unroll
                    for (int j = 0; j < K_; ++j) {
                        if (dv < bd[r][j]) {
                            const float td = bd[r][j]; const int ti = bi[r][j];
                            bd[r][j] = dv; bi[r][j] = di; dv = td; di = ti;
                        }
                    }
                }
            }
        }

        brow += 16 * D_;                        // next 16 columns
    }

    // dump per-lane candidates
    {
        const int base = ((wave * 32 + lane) * 8) * K_;
        #pragma unroll
        for (int r = 0; r < 8; ++r)
            #pragma unroll
            for (int j = 0; j < K_; ++j) {
                md[base + r * K_ + j] = bd[r][j];
                mi[base + r * K_ + j] = bi[r][j];
            }
    }
    __syncthreads();

    // per-row merge: row m<8 lives in lanes 0-15 (reg r=m); m>=8 in lanes 16-31
    if (lane < 16) {
        const int m = lane;
        const int r = (m < 8) ? m : (m - 8);
        const int lstart = (m < 8) ? 0 : 16;
        float fd[K_]; int fi[K_];
        #pragma unroll
        for (int j = 0; j < K_; ++j) { fd[j] = BIGF; fi[j] = 0; }
        for (int l = 0; l < 16; ++l) {
            const int src = ((wave * 32 + lstart + l) * 8 + r) * K_;
            #pragma unroll
            for (int j = 0; j < K_; ++j) {
                float dv = md[src + j];
                int   di = mi[src + j];
                #pragma unroll
                for (int jj = 0; jj < K_; ++jj) {
                    if (dv < fd[jj]) {
                        const float td = fd[jj]; const int ti = fi[jj];
                        fd[jj] = dv; fi[jj] = di; dv = td; di = ti;
                    }
                }
            }
        }
        const int slot = (mtile * 25 + chunk) * 4 + wave;  // = mtile*100 + chunk*4 + wave
        #pragma unroll
        for (int j = 0; j < K_; ++j) {
            pd[(long)slot * (16 * K_) + m * K_ + j] = fd[j];
            pi[(long)slot * (16 * K_) + m * K_ + j] = fi[j];
        }
    }
}

// ---------------------------------------------------------------------------
// 8) per-row merge of 100 sorted 5-lists -> final top-5 -> blend Q
//    grid = 512 blocks (one per batch row), block = 128
// ---------------------------------------------------------------------------
__global__ __launch_bounds__(128) void merge_blend_kernel(
    const float* __restrict__ pd, const int* __restrict__ pi,
    const float* __restrict__ q, const float* __restrict__ buf_q,
    float* __restrict__ out)
{
    __shared__ float sd[128 * K_];
    __shared__ int   si[128 * K_];
    const int b = blockIdx.x;
    const int t = threadIdx.x;
    const int mtile = b >> 4;
    const int m = b & 15;

    float fd[K_]; int fi[K_];
    #pragma unroll
    for (int j = 0; j < K_; ++j) { fd[j] = BIGF; fi[j] = 0; }
    if (t < 100) {
        const int slot = mtile * 100 + t;
        const float* p1 = pd + (long)slot * (16 * K_) + m * K_;
        const int*   p2 = pi + (long)slot * (16 * K_) + m * K_;
        #pragma unroll
        for (int j = 0; j < K_; ++j) { fd[j] = p1[j]; fi[j] = p2[j]; }
    }
    #pragma unroll
    for (int j = 0; j < K_; ++j) { sd[t * K_ + j] = fd[j]; si[t * K_ + j] = fi[j]; }
    __syncthreads();

    for (int s = 64; s >= 1; s >>= 1) {
        if (t < s) {
            const int o = t + s;
            #pragma unroll
            for (int j = 0; j < K_; ++j) {
                float dv = sd[o * K_ + j];
                int   di = si[o * K_ + j];
                #pragma unroll
                for (int jj = 0; jj < K_; ++jj) {
                    if (dv < fd[jj]) {
                        const float td = fd[jj]; const int ti = fi[jj];
                        fd[jj] = dv; fi[jj] = di; dv = td; di = ti;
                    }
                }
            }
            #pragma unroll
            for (int j = 0; j < K_; ++j) { sd[t * K_ + j] = fd[j]; si[t * K_ + j] = fi[j]; }
        }
        __syncthreads();
    }

    // final top-5 indices in si[0..4]
    if (t < A_) {
        float s = 0.0f;
        #pragma unroll
        for (int k = 0; k < K_; ++k)
            s += buf_q[(long)si[k] * A_ + t];
        const float np = s * (1.0f / (float)K_);
        out[b * A_ + t] = 0.5f * q[b * A_ + t] + 0.5f * np;
    }
}

// ---------------------------------------------------------------------------
// launch
// ---------------------------------------------------------------------------
extern "C" void kernel_launch(void* const* d_in, const int* in_sizes, int n_in,
                              void* d_out, int out_size, void* d_ws, size_t ws_size,
                              hipStream_t stream)
{
    const float* x    = (const float*)d_in[0];
    const float* c1w  = (const float*)d_in[1];
    const float* c1b  = (const float*)d_in[2];
    const float* c2w  = (const float*)d_in[3];
    const float* c2b  = (const float*)d_in[4];
    const float* c3w  = (const float*)d_in[5];
    const float* c3b  = (const float*)d_in[6];
    const float* fcw  = (const float*)d_in[7];
    const float* fcb  = (const float*)d_in[8];
    const float* ow   = (const float*)d_in[9];
    const float* ob   = (const float*)d_in[10];
    const float* bemb = (const float*)d_in[11];
    const float* bq   = (const float*)d_in[12];

    float* out_q = (float*)d_out;                  // [512,18]
    float* out_h = (float*)d_out + B_ * A_;        // [512,256]

    char* ws = (char*)d_ws;
    size_t off = 0;
    auto carve = [&](size_t bytes) -> void* {
        void* p = ws + off;
        off = (off + bytes + 255) & ~(size_t)255;
        return p;
    };

    float*  h1  = (float*)carve((size_t)512 * 32 * 20 * 20 * 4);  // 26.2 MB
    float*  h2  = (float*)carve((size_t)512 * 64 * 9 * 9 * 4);    // 10.6 MB
    float*  h3  = (float*)carve((size_t)512 * 3136 * 4);          // 6.4 MB
    float*  hf  = (float*)carve((size_t)B_ * D_ * 4);
    __bf16* hb  = (__bf16*)carve((size_t)B_ * D_ * 2);
    __bf16* eb  = (__bf16*)carve((size_t)CAPP_ * D_ * 2);         // 26.2 MB (padded)
    float*  bn  = (float*)carve((size_t)CAPP_ * 4);               // padded
    float*  qp  = (float*)carve((size_t)B_ * A_ * 4);
    float*  pdv = (float*)carve((size_t)3200 * 16 * K_ * 4);      // 32*25*4 slots
    int*    piv = (int*)carve((size_t)3200 * 16 * K_ * 4);

    emb_convert_kernel<<<CAP_, 256, 0, stream>>>(bemb, eb, bn);

    conv1_kernel<<<(512 * 32 * 20 * 20 + 255) / 256, 256, 0, stream>>>(x, c1w, c1b, h1);
    conv2_kernel<<<(512 * 64 * 9 * 9 + 255) / 256, 256, 0, stream>>>(h1, c2w, c2b, h2);
    conv3_kernel<<<(512 * 64 * 7 * 7 + 255) / 256, 256, 0, stream>>>(h2, c3w, c3b, h3);

    fc_kernel<<<(B_ * D_) / 256, 256, 0, stream>>>(h3, fcw, fcb, hf, hb, out_h);
    qpar_kernel<<<(B_ * A_ + 255) / 256, 256, 0, stream>>>(hf, ow, ob, qp);

    dist_topk_kernel<<<dim3(32, 25), 128, 0, stream>>>(hb, eb, bn, pdv, piv);
    merge_blend_kernel<<<B_, 128, 0, stream>>>(pdv, piv, qp, bq, out_q);
}
